// MultiHeadAttention_59785944760902
// MI455X (gfx1250) — matrix operand order repaired and checked by reading, not compiled, not run on hardware
//
#include <hip/hip_runtime.h>
#include <hip/hip_bf16.h>

typedef __attribute__((ext_vector_type(16))) _Float16 v16h;
typedef __attribute__((ext_vector_type(8)))  _Float16 v8h;
typedef __attribute__((ext_vector_type(8)))  float    v8f;

#define NH   16
#define DH   64
#define NQ_  2048
#define NKV_ 2048
#define NB   2
#define DM   1024   // d_model == inner

#if __has_builtin(__builtin_amdgcn_global_load_async_to_lds_b128)
#define HAS_ASYNC_LDS 1
typedef int v4i_b128 __attribute__((vector_size(16)));
typedef __attribute__((address_space(1))) v4i_b128* gptr_b128;
typedef __attribute__((address_space(3))) v4i_b128* lptr_b128;
#else
#define HAS_ASYNC_LDS 0
#endif

// 16-byte global -> LDS copy; async DMA path on gfx1250 when available.
static __device__ __forceinline__ void cp_async16(_Float16* ldst, const _Float16* gsrc) {
#if HAS_ASYNC_LDS
  __builtin_amdgcn_global_load_async_to_lds_b128((gptr_b128)gsrc, (lptr_b128)ldst, 0, 0);
#else
  *(uint4*)ldst = *(const uint4*)gsrc;
#endif
}

static __device__ __forceinline__ void cp_async_wait() {
#if HAS_ASYNC_LDS
#if __has_builtin(__builtin_amdgcn_s_wait_asynccnt)
  __builtin_amdgcn_s_wait_asynccnt(0);
#else
  asm volatile("s_wait_asynccnt 0" ::: "memory");
#endif
#endif
}

static __device__ __forceinline__ v8f vzero8() {
  v8f z = {0.f, 0.f, 0.f, 0.f, 0.f, 0.f, 0.f, 0.f};
  return z;
}

static __device__ __forceinline__ v16h make_frag(v8h lo, v8h hi) {
  v16h r;
#pragma unroll
  for (int i = 0; i < 8; ++i) { r[i] = lo[i]; r[i + 8] = hi[i]; }
  return r;
}

// A fragment 16x32 f16 from row-major source (stride in halves, multiple of 8).
// ISA layout: lane m=lane&15, khalf=lane>>4; holds K = khalf*8+0..7 and 16+khalf*8+0..7.
static __device__ __forceinline__ v16h load_a_frag(const _Float16* base, int stride) {
  const int lane = threadIdx.x & 31;
  const int m = lane & 15, kh = lane >> 4;
  const _Float16* p = base + m * stride + kh * 8;
  v8h lo = *(const v8h*)p;
  v8h hi = *(const v8h*)(p + 16);
  return make_frag(lo, hi);
}

// B fragment 32x16 f16 from Bt (i.e. column-major B) row-major [N][K] (stride mult of 8).
// ISA layout: lane n=lane&15, khalf=lane>>4; holds K = khalf*8+0..7 and 16+khalf*8+0..7 of column n.
static __device__ __forceinline__ v16h load_b_frag(const _Float16* base, int stride) {
  const int lane = threadIdx.x & 31;
  const int n = lane & 15, kh = lane >> 4;
  const _Float16* p = base + n * stride + kh * 8;
  v8h lo = *(const v8h*)p;
  v8h hi = *(const v8h*)(p + 16);
  return make_frag(lo, hi);
}

static __device__ __forceinline__ v8f wmma_f16(v16h a, v16h b, v8f c) {
  return __builtin_amdgcn_wmma_f32_16x16x32_f16(false, a, false, b, (short)0, c,
                                                false, false);
}

// ---------------------------------------------------------------------------
// Kernel 1: fused Q/K/V projection GEMM.  z = 0:Q, 1:K, 2:V.
// C[4096x1024] = X[4096x1024] * W[1024x1024] + bias, stored f16 head-major.
// V is stored transposed [b][h][d][kv] so it is the Bt operand of P*V.
// ---------------------------------------------------------------------------
__global__ __launch_bounds__(128)
void mha_proj_kernel(const float* __restrict__ Qin, const float* __restrict__ Kin,
                     const float* __restrict__ Vin,
                     const float* __restrict__ Wq, const float* __restrict__ bq,
                     const float* __restrict__ Wk, const float* __restrict__ bk,
                     const float* __restrict__ Wv, const float* __restrict__ bvp,
                     _Float16* __restrict__ qh, _Float16* __restrict__ khd,
                     _Float16* __restrict__ vT) {
  __shared__ alignas(16) _Float16 At[64][40];
  __shared__ alignas(16) _Float16 Bt[64][40];

  const int z = blockIdx.z;
  const float* X    = (z == 0) ? Qin : (z == 1) ? Kin : Vin;
  const float* W    = (z == 0) ? Wq  : (z == 1) ? Wk  : Wv;
  const float* bias = (z == 0) ? bq  : (z == 1) ? bk  : bvp;

  const int n0  = blockIdx.x * 64;
  const int gm0 = blockIdx.y * 64;
  const int t   = threadIdx.x;
  const int wv  = t >> 5;
  const int lane = t & 31, cl = lane & 15, hf = lane >> 4;

  v8f acc[4];
#pragma unroll
  for (int j = 0; j < 4; ++j) acc[j] = vzero8();

  for (int k0 = 0; k0 < DM; k0 += 32) {
    if (k0 + 32 < DM) {
      __builtin_prefetch(W + (size_t)(k0 + 32) * DM + n0, 0, 1);
      __builtin_prefetch(X + (size_t)gm0 * DM + k0 + 32, 0, 1);
    }
    // Stage A tile 64x32, fp32 -> f16 (conversion: must go through VALU)
#pragma unroll
    for (int i = 0; i < 4; ++i) {
      const int idx = t + i * 128;          // 512 float4 chunks
      const int row = idx >> 3, c4 = idx & 7;
      const float4 v = *(const float4*)(X + (size_t)(gm0 + row) * DM + k0 + c4 * 4);
      _Float16* d = &At[row][c4 * 4];
      d[0] = (_Float16)v.x; d[1] = (_Float16)v.y;
      d[2] = (_Float16)v.z; d[3] = (_Float16)v.w;
    }
    // Stage W transposed: Bt[n][kk] = W[k0+kk][n0+n]
#pragma unroll
    for (int i = 0; i < 16; ++i) {
      const int idx = t + i * 128;          // 2048 elements
      const int kk = idx >> 6, n = idx & 63;
      Bt[n][kk] = (_Float16)W[(size_t)(k0 + kk) * DM + n0 + n];
    }
    __syncthreads();
    const v16h a = load_a_frag(&At[wv * 16][0], 40);
#pragma unroll
    for (int j = 0; j < 4; ++j) {
      const v16h b = load_b_frag(&Bt[j * 16][0], 40);
      acc[j] = wmma_f16(a, b, acc[j]);
    }
    __syncthreads();
  }

  // Epilogue: bias add, convert to f16, scatter to head-major layouts.
#pragma unroll
  for (int j = 0; j < 4; ++j) {
    const int n = n0 + j * 16 + cl;
    const float bb = bias[n];
    const int h = n >> 6, d = n & 63;
#pragma unroll
    for (int r = 0; r < 8; ++r) {
      const int gm = gm0 + wv * 16 + r + 8 * hf;     // 0..4095
      const int bi = gm >> 11, m = gm & (NQ_ - 1);
      const float val = acc[j][r] + bb;
      if (z == 2) {
        vT[(((size_t)bi * NH + h) * DH + d) * NKV_ + m] = (_Float16)val;
      } else {
        _Float16* o = (z == 0) ? qh : khd;
        o[(((size_t)bi * NH + h) * NQ_ + m) * DH + d] = (_Float16)val;
      }
    }
  }
}

// ---------------------------------------------------------------------------
// Kernel 2: flash attention.  One (b,h, 64-row q tile) per 128-thread block.
// Each wave owns 16 q rows; kv swept in chunks of 64 with online softmax.
// K/V chunk staging uses async global->LDS DMA when available.
// ---------------------------------------------------------------------------
__global__ __launch_bounds__(128)
void mha_attn_kernel(const _Float16* __restrict__ qh, const _Float16* __restrict__ khd,
                     const _Float16* __restrict__ vT, const int* __restrict__ mask,
                     _Float16* __restrict__ ctx) {
  __shared__ alignas(16) _Float16 qtile[64][72];
  __shared__ alignas(16) _Float16 kbuf[64][72];
  __shared__ alignas(16) _Float16 vbuf[64][72];
  __shared__ alignas(16) _Float16 pbuf[4][16][72];

  const int q0 = blockIdx.x * 64;
  const int bh = blockIdx.y;
  const int b = bh >> 4, h = bh & 15;
  const int t = threadIdx.x, wv = t >> 5;
  const int lane = t & 31, cl = lane & 15, hf = lane >> 4;

  const _Float16* qbase = qh  + ((size_t)bh * NQ_ + q0) * DH;
  const _Float16* kbase = khd + (size_t)bh * NKV_ * DH;
  const _Float16* vbase = vT  + (size_t)bh * DH * NKV_;

  // Stage q tile (f16, 64x64), keep fragments in registers for whole kernel.
#pragma unroll
  for (int i = 0; i < 4; ++i) {
    const int idx = t + i * 128;
    const int row = idx >> 3, c8 = idx & 7;
    cp_async16(&qtile[row][c8 * 8], qbase + (size_t)row * DH + c8 * 8);
  }
  cp_async_wait();
  __syncthreads();
  const v16h aq0 = load_a_frag(&qtile[wv * 16][0], 72);
  const v16h aq1 = load_a_frag(&qtile[wv * 16][32], 72);

  v8f accO[4];
#pragma unroll
  for (int j = 0; j < 4; ++j) accO[j] = vzero8();
  float mrow[8], lrow[8];
#pragma unroll
  for (int r = 0; r < 8; ++r) { mrow[r] = -INFINITY; lrow[r] = 0.f; }

  const float scale = 0.125f;  // 1/sqrt(64)

  for (int c0 = 0; c0 < NKV_; c0 += 64) {
    __syncthreads();   // previous chunk's consumers done with kbuf/vbuf
#pragma unroll
    for (int i = 0; i < 4; ++i) {
      const int idx = t + i * 128;
      const int row = idx >> 3, c8 = idx & 7;
      cp_async16(&kbuf[row][c8 * 8], kbase + (size_t)(c0 + row) * DH + c8 * 8);
      cp_async16(&vbuf[row][c8 * 8], vbase + (size_t)row * NKV_ + c0 + c8 * 8);
    }
    cp_async_wait();
    __syncthreads();

    // S = q * k^T  (4 column tiles of 16, K=64 in two WMMA steps)
    v8f S[4];
#pragma unroll
    for (int j = 0; j < 4; ++j) {
      v8f s = vzero8();
      s = wmma_f16(aq0, load_b_frag(&kbuf[j * 16][0], 72), s);
      s = wmma_f16(aq1, load_b_frag(&kbuf[j * 16][32], 72), s);
      S[j] = s;
    }
    // scale + mask (C layout: element (M = r+8*hf, N = cl))
#pragma unroll
    for (int j = 0; j < 4; ++j) {
#pragma unroll
      for (int r = 0; r < 8; ++r) {
        const int qrow = q0 + wv * 16 + r + 8 * hf;
        const int col = c0 + j * 16 + cl;
        const int mk = mask[((size_t)b * NQ_ + qrow) * NKV_ + col];
        S[j][r] = mk ? S[j][r] * scale : -1e9f;
      }
    }
    // Online softmax: row M = r + 8*hf lives in VGPR slot r across the 16
    // lanes of this half -> reduce with shfl_xor {1,2,4,8} (stays in half).
#pragma unroll
    for (int r = 0; r < 8; ++r) {
      float mx = fmaxf(fmaxf(S[0][r], S[1][r]), fmaxf(S[2][r], S[3][r]));
#pragma unroll
      for (int x = 1; x <= 8; x <<= 1) mx = fmaxf(mx, __shfl_xor(mx, x, 32));
      const float mnew = fmaxf(mrow[r], mx);
      const float alpha = __expf(mrow[r] - mnew);
      float ps = 0.f;
#pragma unroll
      for (int j = 0; j < 4; ++j) { S[j][r] = __expf(S[j][r] - mnew); ps += S[j][r]; }
#pragma unroll
      for (int x = 1; x <= 8; x <<= 1) ps += __shfl_xor(ps, x, 32);
      lrow[r] = lrow[r] * alpha + ps;
      mrow[r] = mnew;
#pragma unroll
      for (int jd = 0; jd < 4; ++jd) accO[jd][r] *= alpha;
    }
    // C-layout P  ->  LDS  ->  A-layout fragments
#pragma unroll
    for (int j = 0; j < 4; ++j)
#pragma unroll
      for (int r = 0; r < 8; ++r)
        pbuf[wv][r + 8 * hf][j * 16 + cl] = (_Float16)S[j][r];
    __syncthreads();

    const v16h ap0 = load_a_frag(&pbuf[wv][0][0], 72);
    const v16h ap1 = load_a_frag(&pbuf[wv][0][32], 72);
#pragma unroll
    for (int jd = 0; jd < 4; ++jd) {
      accO[jd] = wmma_f16(ap0, load_b_frag(&vbuf[jd * 16][0], 72), accO[jd]);
      accO[jd] = wmma_f16(ap1, load_b_frag(&vbuf[jd * 16][32], 72), accO[jd]);
    }
  }

  // Epilogue: normalize and write context [b][n][h*64+d] as f16.
#pragma unroll
  for (int jd = 0; jd < 4; ++jd) {
#pragma unroll
    for (int r = 0; r < 8; ++r) {
      const int qrow = q0 + wv * 16 + r + 8 * hf;
      const float inv = 1.0f / lrow[r];
      ctx[((size_t)b * NQ_ + qrow) * DM + h * DH + jd * 16 + cl] =
          (_Float16)(accO[jd][r] * inv);
    }
  }
}

// ---------------------------------------------------------------------------
// Kernel 3: output projection. out[4096x1024] = ctx(f16) * Wo + bo (fp32 out).
// A tile staged with async global->LDS DMA when available.
// ---------------------------------------------------------------------------
__global__ __launch_bounds__(128)
void mha_oproj_kernel(const _Float16* __restrict__ ctx, const float* __restrict__ Wo,
                      const float* __restrict__ bo, float* __restrict__ out) {
  __shared__ alignas(16) _Float16 At[64][40];
  __shared__ alignas(16) _Float16 Bt[64][40];

  const int n0  = blockIdx.x * 64;
  const int gm0 = blockIdx.y * 64;
  const int t = threadIdx.x, wv = t >> 5;
  const int lane = t & 31, cl = lane & 15, hf = lane >> 4;

  v8f acc[4];
#pragma unroll
  for (int j = 0; j < 4; ++j) acc[j] = vzero8();

  for (int k0 = 0; k0 < DM; k0 += 32) {
    if (k0 + 32 < DM) __builtin_prefetch(Wo + (size_t)(k0 + 32) * DM + n0, 0, 1);
    // Stage A tile (already f16): 64x32 halves, 16B chunks
#pragma unroll
    for (int i = 0; i < 2; ++i) {
      const int idx = t + i * 128;          // 256 chunks
      const int row = idx >> 2, c8 = idx & 3;
      cp_async16(&At[row][c8 * 8], ctx + (size_t)(gm0 + row) * DM + k0 + c8 * 8);
    }
#pragma unroll
    for (int i = 0; i < 16; ++i) {
      const int idx = t + i * 128;
      const int kk = idx >> 6, n = idx & 63;
      Bt[n][kk] = (_Float16)Wo[(size_t)(k0 + kk) * DM + n0 + n];
    }
    cp_async_wait();
    __syncthreads();
    const v16h a = load_a_frag(&At[wv * 16][0], 40);
#pragma unroll
    for (int j = 0; j < 4; ++j) {
      const v16h b = load_b_frag(&Bt[j * 16][0], 40);
      acc[j] = wmma_f16(a, b, acc[j]);
    }
    __syncthreads();
  }

#pragma unroll
  for (int j = 0; j < 4; ++j) {
    const int n = n0 + j * 16 + cl;
    const float bb = bo[n];
#pragma unroll
    for (int r = 0; r < 8; ++r) {
      const int gm = gm0 + wv * 16 + r + 8 * hf;
      out[(size_t)gm * DM + n] = acc[j][r] + bb;
    }
  }
}

extern "C" void kernel_launch(void* const* d_in, const int* in_sizes, int n_in,
                              void* d_out, int out_size, void* d_ws, size_t ws_size,
                              hipStream_t stream) {
  const float* Q    = (const float*)d_in[0];
  const float* K    = (const float*)d_in[1];
  const float* V    = (const float*)d_in[2];
  const int*   mask = (const int*)d_in[3];
  const float* Wq = (const float*)d_in[4];  const float* bq = (const float*)d_in[5];
  const float* Wk = (const float*)d_in[6];  const float* bk = (const float*)d_in[7];
  const float* Wv = (const float*)d_in[8];  const float* bv = (const float*)d_in[9];
  const float* Wo = (const float*)d_in[10]; const float* bo = (const float*)d_in[11];
  float* out = (float*)d_out;

  // f16 workspace: qh | kh | vT | ctx  (8 MB each, 32 MB total)
  _Float16* ws  = (_Float16*)d_ws;
  const size_t per = (size_t)NB * NH * NQ_ * DH;   // 4 Mi halves
  _Float16* qh  = ws;
  _Float16* khd = ws + per;
  _Float16* vT  = ws + 2 * per;
  _Float16* ctx = ws + 3 * per;

  dim3 blk(128);
  mha_proj_kernel<<<dim3(DM / 64, (NB * NQ_) / 64, 3), blk, 0, stream>>>(
      Q, K, V, Wq, bq, Wk, bk, Wv, bv, qh, khd, vT);
  mha_attn_kernel<<<dim3(NQ_ / 64, NB * NH), blk, 0, stream>>>(
      qh, khd, vT, mask, ctx);
  mha_oproj_kernel<<<dim3(DM / 64, (NB * NQ_) / 64), blk, 0, stream>>>(
      ctx, Wo, bo, out);
}
